// class_predictor_51539608233
// MI455X (gfx1250) — compile-verified, matrix operand order checked by reading, not run on hardware
//
#include <hip/hip_runtime.h>
#include <hip/hip_bf16.h>

#define EPS 1e-5f

typedef __bf16 v16bf __attribute__((ext_vector_type(16)));
typedef float  v8f   __attribute__((ext_vector_type(8)));

// ---------------------------------------------------------------------------
// Kernel 1: classifier. One wave32 per token.
//   h1 = relu(BN(x@W1+b1)) [128], h2 = relu(BN(h1@W2+b2)) [32],
//   z = sigmoid(h2@W3+b3), idx = clip(rint(z),0,1)
// ---------------------------------------------------------------------------
__global__ void classifier_kernel(const float* __restrict__ x,
                                  const float* __restrict__ W1, const float* __restrict__ b1,
                                  const float* __restrict__ g1, const float* __restrict__ be1,
                                  const float* __restrict__ m1, const float* __restrict__ v1,
                                  const float* __restrict__ W2, const float* __restrict__ b2,
                                  const float* __restrict__ g2, const float* __restrict__ be2,
                                  const float* __restrict__ m2, const float* __restrict__ v2,
                                  const float* __restrict__ W3, const float* __restrict__ b3,
                                  int* __restrict__ idx)
{
    __shared__ float xs[8][512];
    __shared__ float hs[8][128];
    const int wave = threadIdx.x >> 5;
    const int lane = threadIdx.x & 31;
    const int t    = blockIdx.x * 8 + wave;          // token id, 0..32767

    // stage the 512-float x row in LDS (each lane: 4 x float4)
    const float4* xr = (const float4*)(x + (size_t)t * 512);
    float4* xd = (float4*)xs[wave];
#pragma unroll
    for (int i = 0; i < 4; ++i) xd[lane * 4 + i] = xr[lane * 4 + i];
    __syncthreads();

    // layer 1: each lane owns hidden units j = lane + 32k, k=0..3
    float acc[4] = {0.f, 0.f, 0.f, 0.f};
    for (int d = 0; d < 512; ++d) {
        const float xv = xs[wave][d];
        const float* wr = W1 + d * 128 + lane;
        acc[0] = fmaf(xv, wr[0],  acc[0]);
        acc[1] = fmaf(xv, wr[32], acc[1]);
        acc[2] = fmaf(xv, wr[64], acc[2]);
        acc[3] = fmaf(xv, wr[96], acc[3]);
    }
#pragma unroll
    for (int k = 0; k < 4; ++k) {
        const int j = lane + 32 * k;
        const float s = g1[j] * rsqrtf(v1[j] + EPS);
        float h = (acc[k] + b1[j] - m1[j]) * s + be1[j];
        hs[wave][j] = fmaxf(h, 0.f);
    }
    __syncthreads();

    // layer 2: lane j = lane owns one of 32 units
    float acc2 = 0.f;
    for (int i = 0; i < 128; ++i)
        acc2 = fmaf(hs[wave][i], W2[i * 32 + lane], acc2);
    const float s2 = g2[lane] * rsqrtf(v2[lane] + EPS);
    float h2 = (acc2 + b2[lane] - m2[lane]) * s2 + be2[lane];
    h2 = fmaxf(h2, 0.f);

    // layer 3: dot with W3 across the wave, then sigmoid/round on lane 0
    float v = h2 * W3[lane];
#pragma unroll
    for (int off = 16; off > 0; off >>= 1) v += __shfl_xor(v, off, 32);
    if (lane == 0) {
        const float z = 1.f / (1.f + expf(-(v + b3[0])));
        int e = (int)rintf(z);              // round-half-to-even, like jnp.round
        e = e < 0 ? 0 : (e > 1 ? 1 : e);
        idx[t] = e;
    }
}

// ---------------------------------------------------------------------------
// Kernel 2: swizzle x (f32) into bf16 WMMA A-fragment layout.
// A-matrix 16x32 bf16 layout (CDNA5 ISA):
//   lane<16 : M=lane,   elems 0..7 -> K=0..7,    elems 8..15 -> K=16..23
//   lane>=16: M=lane-16,elems 0..7 -> K=8..15,   elems 8..15 -> K=24..31
// dst element index = ((rt*16 + kc)*512) + lane*16 + e
// ---------------------------------------------------------------------------
__global__ void xswz_kernel(const float* __restrict__ x, __bf16* __restrict__ xsw)
{
    const int gid  = blockIdx.x * 256 + threadIdx.x;   // 2048*16*32 = 1,048,576
    const int lane = gid & 31;
    const int kc   = (gid >> 5) & 15;
    const int rt   = gid >> 9;
    const int M    = lane & 15;
    const int half = lane >> 4;

    const float* src = x + ((size_t)rt * 16 + M) * 512 + kc * 32;
    v16bf v;
#pragma unroll
    for (int e = 0; e < 8; ++e)  v[e]     = (__bf16)src[half * 8 + e];
#pragma unroll
    for (int e = 0; e < 8; ++e)  v[8 + e] = (__bf16)src[16 + half * 8 + e];
    *(v16bf*)(xsw + ((size_t)(rt * 16 + kc)) * 512 + lane * 16) = v;
}

// ---------------------------------------------------------------------------
// Kernel 3: swizzle W_exp (f32 [2,512,720]) into bf16 WMMA B-fragment layout
// (mirror of A: N striped across lanes, same K mapping per element).
// dst element index = (((e*45 + ct)*16 + kc)*512) + lane*16 + el
// ---------------------------------------------------------------------------
__global__ void wswz_kernel(const float* __restrict__ Wexp, __bf16* __restrict__ wsw)
{
    const int gid  = blockIdx.x * 256 + threadIdx.x;   // 2*45*16*32 = 46,080
    if (gid >= 2 * 45 * 16 * 32) return;
    const int lane = gid & 31;
    const int kc   = (gid >> 5) & 15;
    const int ct   = (gid >> 9) % 45;
    const int ex   = gid / (45 * 16 * 32);
    const int N    = lane & 15;
    const int half = lane >> 4;

    v16bf v;
#pragma unroll
    for (int el = 0; el < 16; ++el) {
        const int K = (el < 8) ? (half * 8 + el) : (16 + half * 8 + (el - 8));
        v[el] = (__bf16)Wexp[((size_t)ex * 512 + kc * 32 + K) * 720 + ct * 16 + N];
    }
    *(v16bf*)(wsw + (((size_t)(ex * 45 + ct) * 16 + kc)) * 512 + lane * 16) = v;
}

// ---------------------------------------------------------------------------
// Kernel 4: WMMA GEMM + per-row expert select + bias.
// Block = 256 threads (8 waves). Wave w owns row-tile rt = blockIdx.x*8+w
// (16 rows); block owns 3 col-tiles (48 of 720 cols).
// The block-shared B tile (2 experts x 3 col-tiles x 16 k-chunks x 1KB =
// 96KB) is staged into LDS ONCE via gfx1250 async VMEM->LDS copies
// (global_load_async_to_lds_b128 / s_wait_asynccnt, inline asm).
// Per k-step: batch-load all 6 B fragments (ds_load_b128 clause) + the A
// fragment, THEN run the 6 v_wmma_f32_16x16x32_bf16 — lets the scheduler
// stagger s_wait_dscnt instead of wait-to-zero before every WMMA.
// Accumulators c[j], j = e*3+ct (matches LDS tile order).
// C/D layout: VGPR r, lane: row = r + 8*(lane>>4), col = lane&15.
// ---------------------------------------------------------------------------
__global__ void moe_gemm_kernel(const __bf16* __restrict__ xsw,
                                const __bf16* __restrict__ wsw,
                                const float* __restrict__ bexp,
                                const int* __restrict__ idx,
                                float* __restrict__ out)
{
    extern __shared__ __bf16 bsh[];             // 2*3*16*512 bf16 = 96 KB
    const int lane   = threadIdx.x & 31;
    const int wave   = threadIdx.x >> 5;
    const int rt     = blockIdx.x * 8 + wave;   // 0..2047
    const int ctbase = blockIdx.y * 3;          // 0,3,...,42

    // ---- stage B tile: two contiguous 48KB spans (expert 0, expert 1) ----
    const __bf16* gsrc0 = wsw + ((size_t)(ctbase)      * 16) * 512;
    const __bf16* gsrc1 = wsw + ((size_t)(45 + ctbase) * 16) * 512;
    {
        // 32-bit LDS byte offset of the dynamic-shared base
        const unsigned ldsbase =
            (unsigned)(size_t)(__bf16 __attribute__((address_space(3)))*)bsh;
#pragma unroll
        for (int i = 0; i < 12; ++i) {
            const unsigned ofs = (threadIdx.x + i * 256) * 16;  // b128 chunks
            const char* g0 = (const char*)gsrc0 + ofs;
            const char* g1 = (const char*)gsrc1 + ofs;
            const unsigned l0 = ldsbase + ofs;
            const unsigned l1 = ldsbase + 49152 + ofs;
            asm volatile("global_load_async_to_lds_b128 %0, %1, off"
                         :: "v"(l0), "v"(g0) : "memory");
            asm volatile("global_load_async_to_lds_b128 %0, %1, off"
                         :: "v"(l1), "v"(g1) : "memory");
        }
        asm volatile("s_wait_asynccnt 0x0" ::: "memory");
    }
    __syncthreads();

    // ---- main WMMA loop ----
    v8f c[6];
#pragma unroll
    for (int j = 0; j < 6; ++j)
        c[j] = (v8f){0.f, 0.f, 0.f, 0.f, 0.f, 0.f, 0.f, 0.f};

    const __bf16* aptr = xsw + (size_t)rt * 16 * 512 + lane * 16;
    const __bf16* bptr = bsh + lane * 16;

#pragma unroll
    for (int ks = 0; ks < 16; ++ks) {
        // batch all fragment loads for this k-step first
        const v16bf a = *(const v16bf*)(aptr + (size_t)ks * 512);
        v16bf b[6];
#pragma unroll
        for (int j = 0; j < 6; ++j)
            b[j] = *(const v16bf*)(bptr + ((size_t)(j * 16 + ks)) * 512);
        // then the 6 matrix ops
#pragma unroll
        for (int j = 0; j < 6; ++j)
            c[j] = __builtin_amdgcn_wmma_f32_16x16x32_bf16(
                false, a, false, b[j], (short)0, c[j], false, false);
    }

    // ---- epilogue: per-row expert selection + bias + store ----
    const int half    = lane >> 4;
    const int col     = lane & 15;
    const int rowbase = rt * 16 + half * 8;

    float bias[3][2];
#pragma unroll
    for (int ct = 0; ct < 3; ++ct) {
        const int p = (ctbase + ct) * 16 + col;
        bias[ct][0] = bexp[p];
        bias[ct][1] = bexp[720 + p];
    }
    int esel[8];
#pragma unroll
    for (int r = 0; r < 8; ++r) esel[r] = idx[rowbase + r];

#pragma unroll
    for (int ct = 0; ct < 3; ++ct) {
        const int p = (ctbase + ct) * 16 + col;
#pragma unroll
        for (int r = 0; r < 8; ++r) {
            const int e = esel[r];
            const float val = (e ? c[3 + ct][r] : c[ct][r]) + bias[ct][e];
            out[(size_t)(rowbase + r) * 720 + p] = val;
        }
    }
}

// ---------------------------------------------------------------------------
extern "C" void kernel_launch(void* const* d_in, const int* in_sizes, int n_in,
                              void* d_out, int out_size, void* d_ws, size_t ws_size,
                              hipStream_t stream)
{
    (void)in_sizes; (void)n_in; (void)out_size; (void)ws_size;
    const float* x    = (const float*)d_in[0];   // [64,512,512]
    const float* Wexp = (const float*)d_in[1];   // [2,512,720]
    const float* bexp = (const float*)d_in[2];   // [2,720]
    const float* W1   = (const float*)d_in[3];
    const float* b1   = (const float*)d_in[4];
    const float* g1   = (const float*)d_in[5];
    const float* be1  = (const float*)d_in[6];
    const float* m1   = (const float*)d_in[7];
    const float* v1   = (const float*)d_in[8];
    const float* W2   = (const float*)d_in[9];
    const float* b2   = (const float*)d_in[10];
    const float* g2   = (const float*)d_in[11];
    const float* be2  = (const float*)d_in[12];
    const float* m2   = (const float*)d_in[13];
    const float* v2   = (const float*)d_in[14];
    const float* W3   = (const float*)d_in[15];
    const float* b3   = (const float*)d_in[16];
    float* out = (float*)d_out;

    // workspace layout
    char* ws = (char*)d_ws;
    int*    idx = (int*)ws;                               // 32768 * 4   = 128 KB
    __bf16* xsw = (__bf16*)(ws + 131072);                 // 32768*512*2 = 32 MB
    __bf16* wsw = (__bf16*)(ws + 131072 + 33554432);      // 737280*2    = 1.44 MB

    // 1) classifier -> expert index per token (32768 tokens, 1 wave each)
    classifier_kernel<<<4096, 256, 0, stream>>>(x, W1, b1, g1, be1, m1, v1,
                                                W2, b2, g2, be2, m2, v2, W3, b3, idx);
    // 2) swizzle x to bf16 A-fragments
    xswz_kernel<<<4096, 256, 0, stream>>>(x, xsw);
    // 3) swizzle W_exp to bf16 B-fragments
    wswz_kernel<<<180, 256, 0, stream>>>(Wexp, wsw);
    // 4) WMMA GEMM with per-row expert select (grid 256 x 15 = 3840 WGs,
    //    96KB dynamic LDS for the block-shared B tile)
    moe_gemm_kernel<<<dim3(256, 15), 256, 98304, stream>>>(xsw, wsw, bexp, idx, out);
}
